// Self_Attn_68461778698308
// MI455X (gfx1250) — compile-verified, matrix operand order checked by reading, not run on hardware
//
#include <hip/hip_runtime.h>

typedef __attribute__((ext_vector_type(16))) _Float16 v16h;
typedef __attribute__((ext_vector_type(8)))  _Float16 v8h;
typedef __attribute__((ext_vector_type(8)))  float    v8f;

#define BB 2
#define CC 64
#define C2 32
#define NN 9216   // 96*96

__device__ inline float fast_exp2(float x) {
#if __has_builtin(__builtin_amdgcn_exp2f)
    return __builtin_amdgcn_exp2f(x);
#else
    return exp2f(x);
#endif
}

// ---------------------------------------------------------------------------
// Kernel 1: fused 1x1 conv projections theta/phi/g, fp32 -> fp16
//   theta_h, phi_h : [B][N][32]  (pixel-major rows -> WMMA tile friendly)
//   g_h            : [B][32][N]  (channel-major -> PV A tile: keys contiguous)
// ---------------------------------------------------------------------------
__global__ __launch_bounds__(256)
void proj_kernel(const float* __restrict__ x,
                 const float* __restrict__ w_theta, const float* __restrict__ b_theta,
                 const float* __restrict__ w_phi,   const float* __restrict__ b_phi,
                 const float* __restrict__ w_g,     const float* __restrict__ b_g,
                 _Float16* __restrict__ theta_h, _Float16* __restrict__ phi_h,
                 _Float16* __restrict__ g_h)
{
    int t = blockIdx.x * blockDim.x + threadIdx.x;
    if (t >= BB * NN) return;
    int b = t / NN, n = t % NN;
    float xv[CC];
    const float* xp = x + (size_t)b * CC * NN + n;
    #pragma unroll
    for (int c = 0; c < CC; ++c) xv[c] = xp[(size_t)c * NN];

    for (int o = 0; o < C2; ++o) {
        float at = b_theta[o], ap = b_phi[o], ag = b_g[o];
        const float* wt = w_theta + o * CC;
        const float* wp = w_phi   + o * CC;
        const float* wg = w_g     + o * CC;
        #pragma unroll 16
        for (int c = 0; c < CC; ++c) {
            at += wt[c] * xv[c];
            ap += wp[c] * xv[c];
            ag += wg[c] * xv[c];
        }
        theta_h[((size_t)b * NN + n) * C2 + o]        = (_Float16)at;
        phi_h  [((size_t)b * NN + n) * C2 + o]        = (_Float16)ap;
        g_h[(size_t)b * C2 * NN + (size_t)o * NN + n] = (_Float16)ag;
    }
}

// ---------------------------------------------------------------------------
// Kernel 2: transposed flash attention, 64-key superblocks, software-pipelined.
// One wave per 16-query tile; query = WMMA column = lane, so softmax stats are
// per-lane scalars (one shfl_xor(16) per superblock to merge key halves).
// S' D-layout == P' B-layout => the P transpose is a free f32->f16 pack.
// Next superblock's 8 tiles are prefetched while the current one computes.
// ---------------------------------------------------------------------------
__device__ inline v16h load_tile16(const _Float16* p0, const _Float16* p1) {
    v8h lo = *reinterpret_cast<const v8h*>(p0);
    v8h hi = *reinterpret_cast<const v8h*>(p1);
    v16h r;
    #pragma unroll
    for (int i = 0; i < 8; ++i) { r[i] = lo[i]; r[i + 8] = hi[i]; }
    return r;
}

__global__ __launch_bounds__(32)
void attn_kernel(const _Float16* __restrict__ theta_h,
                 const _Float16* __restrict__ phi_h,
                 const _Float16* __restrict__ g_h,
                 float* __restrict__ attn_o)
{
    const float LOG2E = 1.4426950408889634f;
    const int lane = threadIdx.x;
    const int m16  = lane & 15;
    const int half = lane >> 4;
    const int b    = blockIdx.y;
    const int q0   = blockIdx.x * 16;

    // theta query tile as B matrix (32 ch x 16 queries), lane = column = query.
    const _Float16* tb = theta_h + ((size_t)b * NN + q0) * C2;
    v16h Bq = load_tile16(tb + m16 * C2 + half * 8,
                          tb + m16 * C2 + 16 + half * 8);

    v8f acc0 = {}, acc1 = {};            // outT accum: ch 0-15 / 16-31 (col=query=lane)
    float row_max = -3.0e38f;            // per-lane = per-query running max
    float row_sum = 0.0f;                // per-lane PARTIAL (this lane's key half)

    const _Float16* pb = phi_h + (size_t)b * NN * C2 + (size_t)m16 * C2;
    const _Float16* g0 = g_h + (size_t)b * C2 * NN + (size_t)m16 * NN;
    const _Float16* g1 = g0 + (size_t)16 * NN;

    // ---- prologue: load superblock k0 = 0 ----
    v16h A0  = load_tile16(pb + half * 8,           pb + 16 + half * 8);
    v16h A1  = load_tile16(pb + 16*C2 + half * 8,   pb + 16*C2 + 16 + half * 8);
    v16h A2  = load_tile16(pb + 32*C2 + half * 8,   pb + 32*C2 + 16 + half * 8);
    v16h A3  = load_tile16(pb + 48*C2 + half * 8,   pb + 48*C2 + 16 + half * 8);
    v16h Ga0 = load_tile16(g0 + half * 8,           g0 + 16 + half * 8);
    v16h Gb0 = load_tile16(g1 + half * 8,           g1 + 16 + half * 8);
    v16h Ga1 = load_tile16(g0 + 32 + half * 8,      g0 + 48 + half * 8);
    v16h Gb1 = load_tile16(g1 + 32 + half * 8,      g1 + 48 + half * 8);

    for (int k0 = 0; k0 < NN; k0 += 64) {
        // ---- prefetch next superblock (wraps to 0 on last iter: harmless) ----
        int kn = (k0 + 64 < NN) ? (k0 + 64) : 0;
        const _Float16* pn = pb + (size_t)kn * C2;
        v16h nA0  = load_tile16(pn + half * 8,          pn + 16 + half * 8);
        v16h nA1  = load_tile16(pn + 16*C2 + half * 8,  pn + 16*C2 + 16 + half * 8);
        v16h nA2  = load_tile16(pn + 32*C2 + half * 8,  pn + 32*C2 + 16 + half * 8);
        v16h nA3  = load_tile16(pn + 48*C2 + half * 8,  pn + 48*C2 + 16 + half * 8);
        v16h nGa0 = load_tile16(g0 + kn + half * 8,      g0 + kn + 16 + half * 8);
        v16h nGb0 = load_tile16(g1 + kn + half * 8,      g1 + kn + 16 + half * 8);
        v16h nGa1 = load_tile16(g0 + kn + 32 + half * 8, g0 + kn + 48 + half * 8);
        v16h nGb1 = load_tile16(g1 + kn + 32 + half * 8, g1 + kn + 48 + half * 8);

        // ---- scores: S'[key][query] over 64 keys (tiles already resident) ----
        v8f S0 = {}, S1 = {}, S2 = {}, S3 = {};
        S0 = __builtin_amdgcn_wmma_f32_16x16x32_f16(false, A0, false, Bq, (short)0, S0, false, false);
        S1 = __builtin_amdgcn_wmma_f32_16x16x32_f16(false, A1, false, Bq, (short)0, S1, false, false);
        S2 = __builtin_amdgcn_wmma_f32_16x16x32_f16(false, A2, false, Bq, (short)0, S2, false, false);
        S3 = __builtin_amdgcn_wmma_f32_16x16x32_f16(false, A3, false, Bq, (short)0, S3, false, false);

        // ---- online softmax over 64 keys, all per-lane ----
        float t = fmaxf(fmaxf(S0[0], S1[0]), fmaxf(S2[0], S3[0]));
        #pragma unroll
        for (int r = 1; r < 8; ++r)
            t = fmaxf(t, fmaxf(fmaxf(S0[r], S1[r]), fmaxf(S2[r], S3[r])));
        t = fmaxf(t, __shfl_xor(t, 16));        // merge the two key-half lanes
        float mnew  = fmaxf(row_max, t);
        float c     = mnew * LOG2E;
        float alpha = fast_exp2(fmaf(row_max, LOG2E, -c));
        row_max = mnew;

        float P0[8], P1[8], P2[8], P3[8], s = 0.0f;
        #pragma unroll
        for (int r = 0; r < 8; ++r) {
            P0[r] = fast_exp2(fmaf(S0[r], LOG2E, -c));
            P1[r] = fast_exp2(fmaf(S1[r], LOG2E, -c));
            P2[r] = fast_exp2(fmaf(S2[r], LOG2E, -c));
            P3[r] = fast_exp2(fmaf(S3[r], LOG2E, -c));
            s += (P0[r] + P1[r]) + (P2[r] + P3[r]);
        }
        row_sum = row_sum * alpha + s;          // lane-partial; merged at the end
        #pragma unroll
        for (int r = 0; r < 8; ++r) { acc0[r] *= alpha; acc1[r] *= alpha; }

        // ---- free transpose: S' D-layout == P' B-layout ----
        v16h B20, B21;
        #pragma unroll
        for (int e = 0; e < 8; ++e) {
            B20[e]     = (_Float16)P0[e];
            B20[e + 8] = (_Float16)P1[e];
            B21[e]     = (_Float16)P2[e];
            B21[e + 8] = (_Float16)P3[e];
        }

        acc0 = __builtin_amdgcn_wmma_f32_16x16x32_f16(false, Ga0, false, B20, (short)0, acc0, false, false);
        acc1 = __builtin_amdgcn_wmma_f32_16x16x32_f16(false, Gb0, false, B20, (short)0, acc1, false, false);
        acc0 = __builtin_amdgcn_wmma_f32_16x16x32_f16(false, Ga1, false, B21, (short)0, acc0, false, false);
        acc1 = __builtin_amdgcn_wmma_f32_16x16x32_f16(false, Gb1, false, B21, (short)0, acc1, false, false);

        // ---- rotate double buffers (register renames, no copies) ----
        A0 = nA0; A1 = nA1; A2 = nA2; A3 = nA3;
        Ga0 = nGa0; Gb0 = nGb0; Ga1 = nGa1; Gb1 = nGb1;
    }

    // Merge the two key-half partial sums (lanes i <-> i+16), normalize, store.
    row_sum += __shfl_xor(row_sum, 16);
    float inv = 1.0f / row_sum;

    #pragma unroll
    for (int r = 0; r < 8; ++r) {
        int ch = r + half * 8;
        attn_o[(size_t)b * C2 * NN + (size_t)ch * NN + (q0 + m16)]        = acc0[r] * inv;
        attn_o[(size_t)b * C2 * NN + (size_t)(16 + ch) * NN + (q0 + m16)] = acc1[r] * inv;
    }
}

// ---------------------------------------------------------------------------
// Kernel 3: epilogue. y = x + w_final @ [x ; (w_W @ attn_g + b_W)] + b_final
// ---------------------------------------------------------------------------
__global__ __launch_bounds__(256)
void epilogue_kernel(const float* __restrict__ x,
                     const float* __restrict__ w_W, const float* __restrict__ b_W,
                     const float* __restrict__ w_final, const float* __restrict__ b_final,
                     const float* __restrict__ attn_o,
                     float* __restrict__ out)
{
    int t = blockIdx.x * blockDim.x + threadIdx.x;
    if (t >= BB * NN) return;
    int b = t / NN, n = t % NN;

    float xv[CC];
    const float* xp = x + (size_t)b * CC * NN + n;
    #pragma unroll
    for (int c = 0; c < CC; ++c) xv[c] = xp[(size_t)c * NN];

    float agv[C2];
    const float* ap = attn_o + (size_t)b * C2 * NN + n;
    #pragma unroll
    for (int c = 0; c < C2; ++c) agv[c] = ap[(size_t)c * NN];

    float ag2[CC];
    for (int o = 0; o < CC; ++o) {
        float a = b_W[o];
        const float* w = w_W + o * C2;
        #pragma unroll
        for (int c = 0; c < C2; ++c) a += w[c] * agv[c];
        ag2[o] = a;
    }
    for (int o = 0; o < CC; ++o) {
        float a = b_final[o];
        const float* w = w_final + o * 2 * CC;
        #pragma unroll 16
        for (int c = 0; c < CC; ++c) a += w[c] * xv[c];
        #pragma unroll 16
        for (int c = 0; c < CC; ++c) a += w[CC + c] * ag2[c];
        out[(size_t)b * CC * NN + (size_t)o * NN + n] = xv[o] + a;
    }
}

// ---------------------------------------------------------------------------
extern "C" void kernel_launch(void* const* d_in, const int* in_sizes, int n_in,
                              void* d_out, int out_size, void* d_ws, size_t ws_size,
                              hipStream_t stream)
{
    const float* x       = (const float*)d_in[0];
    const float* w_theta = (const float*)d_in[1];
    const float* b_theta = (const float*)d_in[2];
    const float* w_phi   = (const float*)d_in[3];
    const float* b_phi   = (const float*)d_in[4];
    const float* w_g     = (const float*)d_in[5];
    const float* b_g     = (const float*)d_in[6];
    const float* w_W     = (const float*)d_in[7];
    const float* b_W     = (const float*)d_in[8];
    const float* w_final = (const float*)d_in[9];
    const float* b_final = (const float*)d_in[10];

    // Workspace carve-out (~5.9 MB total):
    _Float16* theta_h = (_Float16*)d_ws;                       // B*N*32 f16
    _Float16* phi_h   = theta_h + (size_t)BB * NN * C2;        // B*N*32 f16
    _Float16* g_h     = phi_h   + (size_t)BB * NN * C2;        // B*32*N f16
    float*    attn_o  = (float*)(g_h + (size_t)BB * C2 * NN);  // B*32*N f32

    proj_kernel<<<(BB * NN + 255) / 256, 256, 0, stream>>>(
        x, w_theta, b_theta, w_phi, b_phi, w_g, b_g, theta_h, phi_h, g_h);

    dim3 ga(NN / 16, BB);   // 576 query tiles x 2 batches, one wave each
    attn_kernel<<<ga, 32, 0, stream>>>(theta_h, phi_h, g_h, attn_o);

    epilogue_kernel<<<(BB * NN + 255) / 256, 256, 0, stream>>>(
        x, w_W, b_W, w_final, b_final, attn_o, (float*)d_out);
}
